// GINGraphClassifier_13606456394542
// MI455X (gfx1250) — matrix operand order repaired and checked by reading, not compiled, not run on hardware
//
#include <hip/hip_runtime.h>
#include <hip/hip_bf16.h>

// ---------------------------------------------------------------------------
// GIN graph classifier for MI455X (gfx1250, wave32).
// Bandwidth-bound workload (~1.8 GB traffic @ 23.3 TB/s => ~80us floor):
//  - per layer: init h=(1+eps)x, scatter-add over edges (coalesced f32 atomics,
//    accumulator L2-resident), then ONE fused kernel doing both 64x64 GEMMs
//    via v_wmma_f32_16x16x32_bf16 (fp32 accum) + ReLU + BN-stat reduction,
//    in-place per 64-row tile. BN finalize + apply are tiny elementwise passes.
//  - activation tile staged global->LDS with GLOBAL_LOAD_ASYNC_TO_LDS_B128
//    (ASYNCcnt) when the toolchain exposes it; sync fallback otherwise.
//  - weights pre-converted once to transposed bf16 so WMMA B-fragments are
//    contiguous 16B loads.
// ---------------------------------------------------------------------------

#define DIM 64
#define TROWS 64
#define BN_EPS 1e-5f

typedef __attribute__((ext_vector_type(8)))  float  v8f;
typedef __attribute__((ext_vector_type(16))) __bf16 v16bf;
typedef __attribute__((ext_vector_type(8)))  __bf16 v8bf;
typedef __attribute__((ext_vector_type(4)))  int    v4i;

// Address-space-qualified pointee types for the async-copy builtin
// (param type per clang diagnostic: "int __vector(4) __device__ *").
typedef __attribute__((address_space(1))) v4i gas_v4i;
typedef __attribute__((address_space(3))) v4i las_v4i;

#if __has_builtin(__builtin_amdgcn_global_load_async_to_lds_b128)
#define HAVE_ASYNC_LDS 1
#else
#define HAVE_ASYNC_LDS 0
#endif

union FragU { v8bf h[2]; v16bf v; };

// Load a 16-element bf16 fragment as two 16B chunks (ds_load_b128 /
// global_load_b128 depending on address space of p0/p1).
__device__ __forceinline__ v16bf ldfrag(const __bf16* p0, const __bf16* p1) {
  FragU f;
  f.h[0] = *reinterpret_cast<const v8bf*>(p0);
  f.h[1] = *reinterpret_cast<const v8bf*>(p1);
  return f.v;
}

// ---------------------------------------------------------------------------
// Weight prep: wT[(l*2+m)][n][k] (bf16) = w_m[l][k][n]   (transposed for WMMA B)
// grid = 6 blocks (3 layers x 2 mats), 256 threads.
// ---------------------------------------------------------------------------
__global__ void prep_weights_kernel(__bf16* __restrict__ wT,
                                    const float* __restrict__ w1,
                                    const float* __restrict__ w2) {
  const int b = blockIdx.x;          // 0..5
  const int l = b >> 1, m = b & 1;
  const float* w = (m == 0 ? w1 : w2) + l * DIM * DIM;
  __bf16* o = wT + b * DIM * DIM;
  for (int i = threadIdx.x; i < DIM * DIM; i += blockDim.x) {
    const int n = i >> 6, k = i & 63;
    o[n * DIM + k] = (__bf16)w[k * DIM + n];
  }
}

// ---------------------------------------------------------------------------
// agg = (1 + eps) * x   (also zeroes the BN stat accumulators)
// ---------------------------------------------------------------------------
__global__ void init_scale_kernel(float* __restrict__ agg,
                                  const float* __restrict__ x,
                                  const float* __restrict__ eps_p,
                                  float* __restrict__ stats,
                                  int total4) {
  if (blockIdx.x == 0 && threadIdx.x < 2 * DIM) stats[threadIdx.x] = 0.0f;
  const int i = blockIdx.x * blockDim.x + threadIdx.x;
  if (i < total4) {
    const float e1 = 1.0f + eps_p[0];
    float4 v = *reinterpret_cast<const float4*>(x + i * 4);
    v.x *= e1; v.y *= e1; v.z *= e1; v.w *= e1;
    *reinterpret_cast<float4*>(agg + i * 4) = v;
  }
}

// ---------------------------------------------------------------------------
// Scatter-add: agg[dst[e]] += x[src[e]].  One wave per edge, 2 floats/lane.
// ---------------------------------------------------------------------------
__global__ void scatter_edges_kernel(float* __restrict__ agg,
                                     const float* __restrict__ x,
                                     const int* __restrict__ src,
                                     const int* __restrict__ dst,
                                     int n_edges) {
  const int e = blockIdx.x * 8 + (threadIdx.x >> 5);
  if (e >= n_edges) return;
  const int lane = threadIdx.x & 31;
  const int c = lane * 2;
  const int s = src[e], d = dst[e];
  const float2 v = *reinterpret_cast<const float2*>(x + (size_t)s * DIM + c);
  atomicAdd(agg + (size_t)d * DIM + c,     v.x);
  atomicAdd(agg + (size_t)d * DIM + c + 1, v.y);
}

// ---------------------------------------------------------------------------
// Fused MLP: h <- relu(relu(h@W1+b1)@W2+b2), in place (64-row tile per block),
// accumulating per-channel sum / sum-of-squares for BatchNorm.
// Block = 256 threads = 8 wave32. Each wave owns (row-tile rt, 2 col-tiles).
// ---------------------------------------------------------------------------
__global__ __launch_bounds__(256) void mlp_kernel(
    float* __restrict__ h,
    const __bf16* __restrict__ wT1, const float* __restrict__ b1,
    const __bf16* __restrict__ wT2, const float* __restrict__ b2,
    float* __restrict__ stats, int nrows) {
#if HAVE_ASYNC_LDS
  __shared__ float Fbuf[TROWS * DIM];            // 16 KB f32 async staging
#endif
  __shared__ __bf16 Abuf[TROWS * DIM];           // 8 KB bf16 activations
  __shared__ __bf16 Hbuf[TROWS * DIM];           // 8 KB bf16 intermediate
  __shared__ float ssum[DIM];
  __shared__ float ssq [DIM];

  const int tid  = threadIdx.x;
  const int row0 = blockIdx.x * TROWS;

  if (tid < DIM) { ssum[tid] = 0.0f; ssq[tid] = 0.0f; }

#if HAVE_ASYNC_LDS
  // ---- stage: contiguous 16KB tile global->LDS via async copy (ASYNCcnt),
  //      then one LDS-resident f32 -> bf16 convert pass. Rows past nrows read
  //      harmless in-workspace garbage; never stored (guarded below).
  {
    const float* gsrc = h + (size_t)row0 * DIM;
#pragma unroll
    for (int j = 0; j < 4; ++j) {
      const int idx = (tid + 256 * j) * 4;       // 16B chunk per call
      __builtin_amdgcn_global_load_async_to_lds_b128(
          (gas_v4i*)(gsrc + idx), (las_v4i*)(&Fbuf[idx]), 0, 0);
    }
#if __has_builtin(__builtin_amdgcn_s_wait_asynccnt)
    __builtin_amdgcn_s_wait_asynccnt(0);
#else
    asm volatile("s_wait_asynccnt 0x0" ::: "memory");
#endif
  }
  __syncthreads();
  {
    const int base = tid * 16;
#pragma unroll
    for (int j = 0; j < 16; ++j) Abuf[base + j] = (__bf16)Fbuf[base + j];
  }
#else
  // ---- sync fallback: 64x64 f32 -> bf16 LDS (zero-pad rows past nrows) ----
  {
    const int r   = tid >> 2;
    const int seg = (tid & 3) * 16;
    const int gr  = row0 + r;
    float4 f0{0,0,0,0}, f1{0,0,0,0}, f2{0,0,0,0}, f3{0,0,0,0};
    if (gr < nrows) {
      const float4* gp =
          reinterpret_cast<const float4*>(h + (size_t)gr * DIM + seg);
      f0 = gp[0]; f1 = gp[1]; f2 = gp[2]; f3 = gp[3];
    }
    v8bf o0, o1;
    o0[0]=(__bf16)f0.x; o0[1]=(__bf16)f0.y; o0[2]=(__bf16)f0.z; o0[3]=(__bf16)f0.w;
    o0[4]=(__bf16)f1.x; o0[5]=(__bf16)f1.y; o0[6]=(__bf16)f1.z; o0[7]=(__bf16)f1.w;
    o1[0]=(__bf16)f2.x; o1[1]=(__bf16)f2.y; o1[2]=(__bf16)f2.z; o1[3]=(__bf16)f2.w;
    o1[4]=(__bf16)f3.x; o1[5]=(__bf16)f3.y; o1[6]=(__bf16)f3.z; o1[7]=(__bf16)f3.w;
    __bf16* dp = Abuf + r * DIM + seg;
    *reinterpret_cast<v8bf*>(dp)     = o0;
    *reinterpret_cast<v8bf*>(dp + 8) = o1;
  }
#endif
  __syncthreads();

  const int wave = tid >> 5;
  const int lane = tid & 31;
  const int rt   = wave & 3;            // row tile 0..3
  const int ctb  = (wave >> 2) * 2;     // first col tile (0 or 2)
  const int half = lane >> 4;
  const int lr   = lane & 15;
  const int arow = rt * 16 + lr;
  const int k0   = half * 8;            // A-fragment K chunk offsets
  const int k1   = 16 + half * 8;
  const int kb   = 16 * half;           // contiguous K base (unused; B uses k0/k1)

  // ---- GEMM1: h1 = relu(A @ W1 + b1) -> Hbuf (bf16) ----
  {
    const __bf16* ap = Abuf + arow * DIM;
    const v16bf a0 = ldfrag(ap + k0,      ap + k1);
    const v16bf a1 = ldfrag(ap + 32 + k0, ap + 32 + k1);
    for (int t = 0; t < 2; ++t) {
      const int n = (ctb + t) * 16 + lr;
      const __bf16* bp = wT1 + n * DIM;
      const v16bf bw0 = ldfrag(bp + k0,      bp + k1);
      const v16bf bw1 = ldfrag(bp + 32 + k0, bp + 32 + k1);
      const float bv = b1[n];
      v8f acc = {bv, bv, bv, bv, bv, bv, bv, bv};
      acc = __builtin_amdgcn_wmma_f32_16x16x32_bf16(
          false, a0, false, bw0, (short)0, acc, false, false);
      acc = __builtin_amdgcn_wmma_f32_16x16x32_bf16(
          false, a1, false, bw1, (short)0, acc, false, false);
#pragma unroll
      for (int i = 0; i < 8; ++i) {
        const int m = rt * 16 + i + 8 * half;
        Hbuf[m * DIM + n] = (__bf16)fmaxf(acc[i], 0.0f);
      }
    }
  }
  __syncthreads();

  // ---- GEMM2: h2 = relu(H @ W2 + b2) -> global h (pre-BN) + stats ----
  {
    const __bf16* ap = Hbuf + arow * DIM;
    const v16bf a0 = ldfrag(ap + k0,      ap + k1);
    const v16bf a1 = ldfrag(ap + 32 + k0, ap + 32 + k1);
    for (int t = 0; t < 2; ++t) {
      const int n = (ctb + t) * 16 + lr;
      const __bf16* bp = wT2 + n * DIM;
      const v16bf bw0 = ldfrag(bp + k0,      bp + k1);
      const v16bf bw1 = ldfrag(bp + 32 + k0, bp + 32 + k1);
      const float bv = b2[n];
      v8f acc = {bv, bv, bv, bv, bv, bv, bv, bv};
      acc = __builtin_amdgcn_wmma_f32_16x16x32_bf16(
          false, a0, false, bw0, (short)0, acc, false, false);
      acc = __builtin_amdgcn_wmma_f32_16x16x32_bf16(
          false, a1, false, bw1, (short)0, acc, false, false);
      float lsum = 0.0f, lsq = 0.0f;
#pragma unroll
      for (int i = 0; i < 8; ++i) {
        const float v = fmaxf(acc[i], 0.0f);
        const int m  = rt * 16 + i + 8 * half;
        const int gr = row0 + m;
        if (gr < nrows) {
          h[(size_t)gr * DIM + n] = v;
          lsum += v; lsq += v * v;
        }
      }
      atomicAdd(&ssum[n], lsum);   // ds_add_f32
      atomicAdd(&ssq[n],  lsq);
    }
  }
  __syncthreads();
  if (tid < DIM) {
    atomicAdd(&stats[tid],        ssum[tid]);
    atomicAdd(&stats[DIM + tid],  ssq[tid]);
  }
}

// ---------------------------------------------------------------------------
// BN finalize: per-channel scale/shift from accumulated sum / sumsq.
// ---------------------------------------------------------------------------
__global__ void bn_finalize_kernel(float* __restrict__ bnp,
                                   const float* __restrict__ stats,
                                   const float* __restrict__ gamma,
                                   const float* __restrict__ beta,
                                   float inv_n) {
  const int c = threadIdx.x;   // 64 threads
  const float mean = stats[c] * inv_n;
  const float var  = stats[DIM + c] * inv_n - mean * mean;
  const float sc   = gamma[c] * rsqrtf(var + BN_EPS);
  bnp[c]        = sc;
  bnp[DIM + c]  = beta[c] - mean * sc;
}

// ---------------------------------------------------------------------------
// out = relu(in * scale[c] + shift[c])
// ---------------------------------------------------------------------------
__global__ void bn_apply_kernel(float* __restrict__ out,
                                const float* __restrict__ in,
                                const float* __restrict__ bnp,
                                int total4) {
  const int i = blockIdx.x * blockDim.x + threadIdx.x;
  if (i >= total4) return;
  const int base = i * 4;
  const int c = base & (DIM - 1);
  float4 v = *reinterpret_cast<const float4*>(in + base);
  v.x = fmaxf(v.x * bnp[c]     + bnp[DIM + c],     0.0f);
  v.y = fmaxf(v.y * bnp[c + 1] + bnp[DIM + c + 1], 0.0f);
  v.z = fmaxf(v.z * bnp[c + 2] + bnp[DIM + c + 2], 0.0f);
  v.w = fmaxf(v.w * bnp[c + 3] + bnp[DIM + c + 3], 0.0f);
  *reinterpret_cast<float4*>(out + base) = v;
}

// ---------------------------------------------------------------------------
// Pooling
// ---------------------------------------------------------------------------
__global__ void zero_pool_kernel(float* __restrict__ pooled,
                                 float* __restrict__ counts,
                                 int n_graphs) {
  const int i = blockIdx.x * blockDim.x + threadIdx.x;
  if (i < n_graphs * DIM) pooled[i] = 0.0f;
  if (i < n_graphs)       counts[i] = 0.0f;
}

__global__ void pool_scatter_kernel(float* __restrict__ pooled,
                                    float* __restrict__ counts,
                                    const float* __restrict__ x,
                                    const int* __restrict__ batch,
                                    int n_nodes) {
  const int gid  = blockIdx.x * blockDim.x + threadIdx.x;
  const int node = gid >> 5;
  if (node >= n_nodes) return;
  const int lane = gid & 31;
  const int c = lane * 2;
  const int g = batch[node];
  const float2 v = *reinterpret_cast<const float2*>(x + (size_t)node * DIM + c);
  atomicAdd(pooled + (size_t)g * DIM + c,     v.x);
  atomicAdd(pooled + (size_t)g * DIM + c + 1, v.y);
  if (lane == 0) atomicAdd(counts + g, 1.0f);
}

__global__ void head_kernel(float* __restrict__ out,
                            const float* __restrict__ pooled,
                            const float* __restrict__ counts,
                            const float* __restrict__ lin_w,
                            const float* __restrict__ lin_b,
                            int n_graphs, int out_dim) {
  const int t = blockIdx.x * blockDim.x + threadIdx.x;
  if (t >= n_graphs * out_dim) return;
  const int g = t / out_dim, o = t % out_dim;
  float s = 0.0f;
#pragma unroll 8
  for (int c = 0; c < DIM; ++c) s += pooled[g * DIM + c] * lin_w[c * out_dim + o];
  float cnt = counts[g]; cnt = cnt > 1.0f ? cnt : 1.0f;
  out[t] = s / cnt + lin_b[o];
}

// ---------------------------------------------------------------------------
// Host-side orchestration (all on `stream`, graph-capture safe)
// ---------------------------------------------------------------------------
extern "C" void kernel_launch(void* const* d_in, const int* in_sizes, int n_in,
                              void* d_out, int out_size, void* d_ws,
                              size_t ws_size, hipStream_t stream) {
  const float* x      = (const float*)d_in[0];
  const int*   eidx   = (const int*)  d_in[1];
  const int*   batch  = (const int*)  d_in[2];
  const float* w1     = (const float*)d_in[3];
  const float* b1     = (const float*)d_in[4];
  const float* w2     = (const float*)d_in[5];
  const float* b2     = (const float*)d_in[6];
  const float* gamma  = (const float*)d_in[7];
  const float* beta   = (const float*)d_in[8];
  const float* eps_g  = (const float*)d_in[9];
  const float* lin_w  = (const float*)d_in[10];
  const float* lin_b  = (const float*)d_in[11];
  float* out = (float*)d_out;

  const int n_nodes  = in_sizes[0] / DIM;      // 100000
  const int n_edges  = in_sizes[1] / 2;        // 1000000
  const int n_graphs = in_sizes[10] ? (out_size / (in_sizes[10] / DIM)) : 256;
  const int out_dim  = in_sizes[11];           // 10
  const int* src = eidx;
  const int* dst = eidx + n_edges;

  // workspace carve-up (all offsets multiples of 256B)
  char* ws = (char*)d_ws;
  const size_t szH = (size_t)n_nodes * DIM * sizeof(float);  // 25.6 MB
  float*   hA     = (float*)(ws);                            // agg / pre-BN
  float*   hB     = (float*)(ws + szH);                      // layer output
  __bf16*  wT     = (__bf16*)(ws + 2 * szH);                 // 6*4096 bf16
  float*   stats  = (float*)(ws + 2 * szH + 49152);          // 128
  float*   bnp    = (float*)(ws + 2 * szH + 49152 + 512);
  float*   pooled = (float*)(ws + 2 * szH + 49152 + 1024);
  float*   counts = (float*)(ws + 2 * szH + 49152 + 1024 + 65536);

  const int total4     = n_nodes * DIM / 4;
  const int elw_blocks = (total4 + 255) / 256;
  const int mlp_blocks = (n_nodes + TROWS - 1) / TROWS;
  const int sct_blocks = (n_edges + 7) / 8;

  prep_weights_kernel<<<6, 256, 0, stream>>>(wT, w1, w2);

  const float* x_cur = x;
  for (int l = 0; l < 3; ++l) {
    init_scale_kernel<<<elw_blocks, 256, 0, stream>>>(hA, x_cur, eps_g + l,
                                                      stats, total4);
    scatter_edges_kernel<<<sct_blocks, 256, 0, stream>>>(hA, x_cur, src, dst,
                                                         n_edges);
    mlp_kernel<<<mlp_blocks, 256, 0, stream>>>(
        hA, wT + (l * 2 + 0) * DIM * DIM, b1 + l * DIM,
        wT + (l * 2 + 1) * DIM * DIM, b2 + l * DIM, stats, n_nodes);
    bn_finalize_kernel<<<1, DIM, 0, stream>>>(bnp, stats, gamma + l * DIM,
                                              beta + l * DIM,
                                              1.0f / (float)n_nodes);
    bn_apply_kernel<<<elw_blocks, 256, 0, stream>>>(hB, hA, bnp, total4);
    x_cur = hB;
  }

  zero_pool_kernel<<<(n_graphs * DIM + 255) / 256 + 1, 256, 0, stream>>>(
      pooled, counts, n_graphs);
  pool_scatter_kernel<<<(n_nodes * 32 + 255) / 256, 256, 0, stream>>>(
      pooled, counts, x_cur, batch, n_nodes);
  head_kernel<<<(n_graphs * out_dim + 255) / 256, 256, 0, stream>>>(
      out, pooled, counts, lin_w, lin_b, n_graphs, out_dim);
}
